// LSTM_17489106829786
// MI455X (gfx1250) — 8-bit matrix operand order repaired and checked by reading, compiled, not run on hardware
//
#include <hip/hip_runtime.h>
#include <hip/hip_bf16.h>

typedef int   v8i_t __attribute__((ext_vector_type(8)));
typedef float v8f_t __attribute__((ext_vector_type(8)));

#define Bn      2048
#define Hn      64
#define Fn      16
#define ORDn    16
#define Kn      256
#define NG      1024
#define KIN     288
#define WSTRIDE 304
#define TSTRIDE 112
#define TAILK   192
#define USTRIDE 296
#define BT      16
#define THREADS 512
#define WSCALE      16.0f
#define INV_WSCALE  0.0625f

__device__ __forceinline__ unsigned char f32_to_e4m3(float f) {
    unsigned int bits = __float_as_uint(f);
    unsigned char sign = (unsigned char)((bits >> 24) & 0x80u);
    float ab = fabsf(f);
    if (ab >= 448.0f) return sign | 0x7E;
    if (ab < 0.000976562f) return sign;
    int e = (int)((bits >> 23) & 0xFF) - 127;
    int q;
    if (e < -6) {
        int m = (int)rintf(ab * 512.0f);
        q = (m >= 8) ? 0x08 : m;
    } else {
        float sc = ab * __uint_as_float((unsigned)(130 - e) << 23);
        int m = (int)rintf(sc);
        if (m >= 16) { m = 8; e += 1; }
        if (e > 8) return sign | 0x7E;
        q = ((e + 7) << 3) | (m - 8);
    }
    return sign | (unsigned char)q;
}

__device__ __forceinline__ float fast_sigmoid(float z) {
    return 1.0f / (1.0f + __expf(-z));
}
__device__ __forceinline__ float fast_tanh(float z) {
    float e = __expf(-2.0f * fabsf(z));
    float t = (1.0f - e) / (1.0f + e);
    return copysignf(t, z);
}

__device__ __forceinline__ v8i_t load_bfrag(const unsigned char* p) {
    int4 q0 = *(const int4*)(p);
    int4 q1 = *(const int4*)(p + 32);
    v8i_t b;
    b[0] = q0.x; b[1] = q0.y; b[2] = q0.z; b[3] = q0.w;
    b[4] = q1.x; b[5] = q1.y; b[6] = q1.z; b[7] = q1.w;
    return b;
}

__global__ void prep_weights_kernel(const float* __restrict__ Wk,
                                    const float* __restrict__ Wr,
                                    unsigned char* __restrict__ dst) {
    int n = blockIdx.x;
    for (int k = threadIdx.x; k < WSTRIDE; k += blockDim.x) {
        unsigned char q = 0;
        if (k < Fn + ORDn)      q = f32_to_e4m3(Wk[k * NG + n] * WSCALE);
        else if (k < KIN)       q = f32_to_e4m3(Wr[(k - (Fn + ORDn)) * NG + n] * WSCALE);
        dst[n * WSTRIDE + k] = q;
    }
}

__global__ void __launch_bounds__(THREADS, 1)
lstm_fp8_kernel(const float* __restrict__ x,
                const float* __restrict__ y0,
                const float* __restrict__ bias,
                const float* __restrict__ Dw,
                const float* __restrict__ Db,
                const unsigned char* __restrict__ wc8,
                float* __restrict__ out) {
    extern __shared__ unsigned char lds[];
    unsigned char* wtail = lds;
    unsigned char* u     = lds + (size_t)NG * TSTRIDE;
    float* yf            = (float*)(u + BT * USTRIDE);
    float* predAcc       = yf + 2 * BT * ORDn;

    const int tid   = threadIdx.x;
    const int wave  = tid >> 5;
    const int lane  = tid & 31;
    const int lcol  = lane & 15;
    const int khalf = lane >> 4;
    const int baseRow = blockIdx.x * BT;

    {
        for (int i = tid; i < NG * 24; i += THREADS) {
            int nc = i / 24, off = (i % 24) * 4;
            *(unsigned int*)(wtail + nc * TSTRIDE + off) =
                *(const unsigned int*)(wc8 + (size_t)nc * WSTRIDE + TAILK + off);
        }
        unsigned int* uw = (unsigned int*)u;
        for (int i = tid; i < BT * USTRIDE / 4; i += THREADS) uw[i] = 0u;
        if (tid < BT) predAcc[tid] = 0.0f;
    }

    const int ncol   = wave * 16 + lcol;
    const float bi   = bias[          ncol];
    const float bf   = bias[Kn      + ncol];
    const float bg   = bias[2 * Kn  + ncol];
    const float bo   = bias[3 * Kn  + ncol];
    const float dwv  = Dw[ncol];
    const float dbv  = Db[0];

    v8i_t breg[3][4];
    #pragma unroll
    for (int kt = 0; kt < 3; ++kt)
        #pragma unroll
        for (int g = 0; g < 4; ++g)
            breg[kt][g] = load_bfrag(
                wc8 + (size_t)(g * Kn + ncol) * WSTRIDE + kt * 64 + khalf * 16);

    __syncthreads();

    float c[8];
    #pragma unroll
    for (int v = 0; v < 8; ++v) c[v] = 0.0f;

    for (int t = 0; t < Hn; ++t) {
        if (tid < BT * ORDn) {
            int m = tid >> 4, j = tid & 15;
            float* ycur = yf + (t & 1) * BT * ORDn;
            float yv;
            if (t == 0) {
                yv = y0[(baseRow + m) * ORDn + j];
            } else {
                if (j == 0) {
                    float pred = predAcc[m] + dbv;
                    out[(size_t)(baseRow + m) * Hn + (t - 1)] = pred;
                    predAcc[m] = 0.0f;
                    yv = pred;
                } else {
                    const float* yprev = yf + ((t - 1) & 1) * BT * ORDn;
                    yv = yprev[m * ORDn + (j - 1)];
                }
            }
            ycur[m * ORDn + j] = yv;
            u[m * USTRIDE + Fn + j] = f32_to_e4m3(yv);
            float xv = x[((size_t)(baseRow + m) * Hn + t) * Fn + j];
            u[m * USTRIDE + j] = f32_to_e4m3(xv);
            if (t + 1 < Hn)
                __builtin_prefetch(&x[((size_t)(baseRow + m) * Hn + t + 1) * Fn + j], 0, 1);
        }
        __syncthreads();

        v8f_t acc[4];
        #pragma unroll
        for (int g = 0; g < 4; ++g)
            #pragma unroll
            for (int v = 0; v < 8; ++v) acc[g][v] = 0.0f;

        const unsigned char* urow = u + lcol * USTRIDE;
        const int ho = khalf * 16;

        #pragma unroll
        for (int kt = 0; kt < 5; ++kt) {
            const int kbase = (kt < 4) ? kt * 64 : 224;
            v8i_t a;
            if (kt < 4) {
                int2 p0 = *(const int2*)(urow + kbase +  0 + ho);
                int2 p1 = *(const int2*)(urow + kbase +  8 + ho);
                a[0] = p0.x; a[1] = p0.y; a[2] = p1.x; a[3] = p1.y;
            } else {
                a[0] = 0; a[1] = 0; a[2] = 0; a[3] = 0;
            }
            int2 p2 = *(const int2*)(urow + kbase + 32 + ho);
            int2 p3 = *(const int2*)(urow + kbase + 40 + ho);
            a[4] = p2.x; a[5] = p2.y; a[6] = p3.x; a[7] = p3.y;

            #pragma unroll
            for (int g = 0; g < 4; ++g) {
                v8i_t bm;
                if (kt < 3) {
                    bm = breg[kt][g];
                } else {
                    bm = load_bfrag(wtail + (size_t)(g * Kn + ncol) * TSTRIDE
                                    + (kbase - TAILK) + khalf * 16);
                }
                acc[g] = __builtin_amdgcn_wmma_f32_16x16x64_fp8_fp8(
                             a, bm, (short)0, acc[g], false, false);
            }
        }

        float hval[8], s[8];
        #pragma unroll
        for (int v = 0; v < 8; ++v) {
            float zi = acc[0][v] * INV_WSCALE + bi;
            float zf = acc[1][v] * INV_WSCALE + bf;
            float zg = acc[2][v] * INV_WSCALE + bg;
            float zo = acc[3][v] * INV_WSCALE + bo;
            float ig = fast_sigmoid(zi);
            float fg = fast_sigmoid(zf);
            float gg = fast_tanh(zg);
            float og = fast_sigmoid(zo);
            c[v] = fg * c[v] + ig * gg;
            float hv = og * fast_tanh(c[v]);
            hval[v] = hv;
            s[v] = hv * dwv;
        }
        __syncthreads();

        #pragma unroll
        for (int v = 0; v < 8; ++v) {
            int m = v + khalf * 8;
            u[m * USTRIDE + (Fn + ORDn) + ncol] = f32_to_e4m3(hval[v]);
            float sv = s[v];
            sv += __shfl_xor(sv, 1, 16);
            sv += __shfl_xor(sv, 2, 16);
            sv += __shfl_xor(sv, 4, 16);
            sv += __shfl_xor(sv, 8, 16);
            if (lcol == 0) atomicAdd(&predAcc[m], sv);
        }
        __syncthreads();
    }

    if (tid < BT) {
        out[(size_t)(baseRow + tid) * Hn + (Hn - 1)] = predAcc[tid] + dbv;
    }
}

extern "C" void kernel_launch(void* const* d_in, const int* in_sizes, int n_in,
                              void* d_out, int out_size, void* d_ws, size_t ws_size,
                              hipStream_t stream) {
    const float* x  = (const float*)d_in[0];
    const float* y0 = (const float*)d_in[1];
    const float* Wk = (const float*)d_in[2];
    const float* Wr = (const float*)d_in[3];
    const float* b  = (const float*)d_in[4];
    const float* Dw = (const float*)d_in[5];
    const float* Db = (const float*)d_in[6];
    float* out = (float*)d_out;
    unsigned char* wc8 = (unsigned char*)d_ws;

    prep_weights_kernel<<<NG, 128, 0, stream>>>(Wk, Wr, wc8);

    size_t shmem = (size_t)NG * TSTRIDE + BT * USTRIDE
                 + 2 * BT * ORDn * sizeof(float) + BT * sizeof(float);
    hipFuncSetAttribute((const void*)lstm_fp8_kernel,
                        hipFuncAttributeMaxDynamicSharedMemorySize, (int)shmem);

    lstm_fp8_kernel<<<Bn / BT, THREADS, shmem, stream>>>(x, y0, b, Dw, Db, wc8, out);
}
